// Model_82317343195670
// MI455X (gfx1250) — compile-verified, matrix operand order checked by reading, not run on hardware
//
#include <hip/hip_runtime.h>
#include <math.h>

// ---------------- model constants ----------------
#define D_MODEL 768
#define FF_DIM  3072
#define NHEAD   12
#define DK      64
#define BATCHN  32
#define NPATCH  49
#define LCAP    60
#define VOCAB   30522
#define SQRT_D  27.712812921102035f
#define LN10K   9.210340371976184f
#define EPS_LN  1e-6f
#define EPS_POS 1e-12f
#define NOMASK  (1 << 30)

typedef __attribute__((ext_vector_type(16))) _Float16 v16h;
typedef __attribute__((ext_vector_type(8)))  float    v8f;
typedef __attribute__((ext_vector_type(4)))  _Float16 h4;

union Frag {
  v16h v;
  uint4 q[2];
  _Float16 h[16];
};

// =================================================================
// Generic GEMM:  C[M,N] = act(A[M,K] @ W[K,N] + bias) (+ residual)
// Block tile 64x128, BK=32, 256 threads = 8 waves; each wave owns a
// 32x32 strip = 4 WMMA accumulators. Double-buffered LDS, software
// pipelined: next tile is loaded to registers while WMMAs run.
// =================================================================
#define BM 64
#define BN 128
#define LP 40   // LDS pitch (halves): 32 + 8 pad -> 80B rows, conflict-free b128 frags

struct StageRegs { float4 a0, a1; float2 b[8]; };

__global__ __launch_bounds__(256) void gemm_kernel(
    const float* __restrict__ A, const float* __restrict__ W,
    const float* __restrict__ bias, const float* __restrict__ res,
    float* __restrict__ C, int M, int N, int K, int act)
{
  __shared__ __align__(16) _Float16 As[2][BM * LP];
  __shared__ __align__(16) _Float16 Bs[2][BN * LP];

  const int tid  = threadIdx.x;
  const int wave = tid >> 5;
  const int lane = tid & 31;
  const int m0 = blockIdx.y * BM;
  const int n0 = blockIdx.x * BN;
  const int wr = (wave >> 2) * 32;   // wave row offset within block (0 / 32)
  const int wc = (wave & 3) * 32;    // wave col offset within block

  // fixed per-thread staging coordinates
  const int ac4 = (tid & 7) * 4;     // A: k-offset of this thread's float4
  const int ar0 = tid >> 3;          // A: first row (second is ar0+32)
  const int bn  = (tid & 63) * 2;    // B: n pair
  const int bk0 = tid >> 6;          // B: k base (k = bk0 + 4j)

  auto load_tile = [&](int k0, StageRegs& r) {
    int gm0 = m0 + ar0, gm1 = gm0 + 32;
    const float* ap0 = A + (size_t)(gm0 < M ? gm0 : M - 1) * K + k0 + ac4;
    const float* ap1 = A + (size_t)(gm1 < M ? gm1 : M - 1) * K + k0 + ac4;
    r.a0 = *(const float4*)ap0;
    r.a1 = *(const float4*)ap1;
    if (gm0 >= M) r.a0 = make_float4(0.f, 0.f, 0.f, 0.f);
    if (gm1 >= M) r.a1 = make_float4(0.f, 0.f, 0.f, 0.f);
    int gn = n0 + bn;
    if (gn + 1 < N) {
#pragma unroll
      for (int j = 0; j < 8; ++j)
        r.b[j] = *(const float2*)&W[(size_t)(k0 + bk0 + 4 * j) * N + gn];
    } else {
#pragma unroll
      for (int j = 0; j < 8; ++j) {
        float x = (gn < N) ? W[(size_t)(k0 + bk0 + 4 * j) * N + gn] : 0.f;
        r.b[j] = make_float2(x, 0.f);
      }
    }
  };

  auto store_tile = [&](int buf, const StageRegs& r) {
    h4 ha;
    ha[0] = (_Float16)r.a0.x; ha[1] = (_Float16)r.a0.y;
    ha[2] = (_Float16)r.a0.z; ha[3] = (_Float16)r.a0.w;
    *(h4*)&As[buf][ar0 * LP + ac4] = ha;
    ha[0] = (_Float16)r.a1.x; ha[1] = (_Float16)r.a1.y;
    ha[2] = (_Float16)r.a1.z; ha[3] = (_Float16)r.a1.w;
    *(h4*)&As[buf][(ar0 + 32) * LP + ac4] = ha;
#pragma unroll
    for (int j = 0; j < 8; ++j) {
      int k = bk0 + 4 * j;
      Bs[buf][bn * LP + k]       = (_Float16)r.b[j].x;
      Bs[buf][(bn + 1) * LP + k] = (_Float16)r.b[j].y;
    }
  };

  v8f acc00 = {}, acc01 = {}, acc10 = {}, acc11 = {};
  const int nk = K >> 5;             // K % 32 == 0 at every call site

  StageRegs rg;
  load_tile(0, rg);
  store_tile(0, rg);
  __syncthreads();

  const int koff = (lane < 16) ? 0 : 8;
  const int la = lane & 15;

  for (int kt = 0; kt < nk; ++kt) {
    const int cur = kt & 1;
    const bool hasNext = (kt + 1 < nk);
    if (hasNext) load_tile((kt + 1) << 5, rg);

    Frag a0, a1, b0, b1;
    a0.q[0] = *(const uint4*)&As[cur][(wr + la) * LP + koff];
    a0.q[1] = *(const uint4*)&As[cur][(wr + la) * LP + koff + 16];
    a1.q[0] = *(const uint4*)&As[cur][(wr + 16 + la) * LP + koff];
    a1.q[1] = *(const uint4*)&As[cur][(wr + 16 + la) * LP + koff + 16];
    b0.q[0] = *(const uint4*)&Bs[cur][(wc + la) * LP + koff];
    b0.q[1] = *(const uint4*)&Bs[cur][(wc + la) * LP + koff + 16];
    b1.q[0] = *(const uint4*)&Bs[cur][(wc + 16 + la) * LP + koff];
    b1.q[1] = *(const uint4*)&Bs[cur][(wc + 16 + la) * LP + koff + 16];

    acc00 = __builtin_amdgcn_wmma_f32_16x16x32_f16(false, a0.v, false, b0.v,
                                                   (short)0, acc00, false, false);
    acc01 = __builtin_amdgcn_wmma_f32_16x16x32_f16(false, a0.v, false, b1.v,
                                                   (short)0, acc01, false, false);
    acc10 = __builtin_amdgcn_wmma_f32_16x16x32_f16(false, a1.v, false, b0.v,
                                                   (short)0, acc10, false, false);
    acc11 = __builtin_amdgcn_wmma_f32_16x16x32_f16(false, a1.v, false, b1.v,
                                                   (short)0, acc11, false, false);

    if (hasNext) store_tile(cur ^ 1, rg);
    __syncthreads();
  }

  auto emit = [&](const v8f& acc, int mt, int nt) {
    int mbase = m0 + wr + mt + ((lane >> 4) << 3);
    int gn = n0 + wc + nt + (lane & 15);
    if (gn >= N) return;
    float bv = bias[gn];
#pragma unroll
    for (int j = 0; j < 8; ++j) {
      int gm = mbase + j;
      if (gm >= M) continue;
      float v = acc[j] + bv;
      if (act == 1)      v = fmaxf(v, 0.0f);
      else if (act == 2) v = 1.0f / (1.0f + expf(-v));
      if (res) v += res[(size_t)gm * N + gn];
      C[(size_t)gm * N + gn] = v;
    }
  };
  emit(acc00, 0, 0);  emit(acc01, 0, 16);
  emit(acc10, 16, 0); emit(acc11, 16, 16);
}

// =================================================================
// Fused attention: one workgroup per (batch, head).
// Q,K,V,O are (B,S,D_MODEL) f32; head h owns cols [h*64, h*64+64).
// LDS: f16 Q, K, V^T, f32 scores, f16 probabilities.
// mask: allowed(i,j) = (j < maskIT) || (j <= i)   (maskIT=NOMASK -> none)
// =================================================================
__global__ __launch_bounds__(256) void attn_kernel(
    const float* __restrict__ Qm, const float* __restrict__ Km,
    const float* __restrict__ Vm, float* __restrict__ Om,
    int S, int Spad, int maskIT, float scale)
{
  extern __shared__ __align__(16) char smem[];
  _Float16* qs = (_Float16*)smem;                  // Spad x 72
  _Float16* ks = qs + (size_t)Spad * 72;           // Spad x 72
  _Float16* vt = ks + (size_t)Spad * 72;           // 64 x (Spad+8)  (V transposed)
  _Float16* ph = vt + (size_t)64 * (Spad + 8);     // Spad x (Spad+8) f16 probs
  float*    sc = (float*)(ph + (size_t)Spad * (Spad + 8)); // Spad x (Spad+4)
  const int scs = Spad + 4;
  const int vts = Spad + 8;
  const int phs = Spad + 8;

  const int tid = threadIdx.x, wave = tid >> 5, lane = tid & 31;
  const int h = blockIdx.x, b = blockIdx.y;
  const size_t base = (size_t)b * S * D_MODEL + (size_t)h * DK;

  // ---- vectorized load + f16 convert (zero-pad rows >= S) ----
  for (int g = tid; g < Spad * 16; g += 256) {
    int s = g >> 4, d4 = (g & 15) * 4;
    const size_t off = base + (size_t)(s < S ? s : 0) * D_MODEL + d4;
    float4 q = *(const float4*)&Qm[off];
    float4 k = *(const float4*)&Km[off];
    float4 v = *(const float4*)&Vm[off];
    if (s >= S) {
      q = make_float4(0.f, 0.f, 0.f, 0.f);
      k = q; v = q;
    }
    h4 hq, hk;
    hq[0] = (_Float16)q.x; hq[1] = (_Float16)q.y; hq[2] = (_Float16)q.z; hq[3] = (_Float16)q.w;
    hk[0] = (_Float16)k.x; hk[1] = (_Float16)k.y; hk[2] = (_Float16)k.z; hk[3] = (_Float16)k.w;
    *(h4*)&qs[s * 72 + d4] = hq;
    *(h4*)&ks[s * 72 + d4] = hk;
    vt[(d4 + 0) * vts + s] = (_Float16)v.x;
    vt[(d4 + 1) * vts + s] = (_Float16)v.y;
    vt[(d4 + 2) * vts + s] = (_Float16)v.z;
    vt[(d4 + 3) * vts + s] = (_Float16)v.w;
  }
  __syncthreads();

  const int nt = Spad >> 4;
  const int koffL = (lane < 16) ? 0 : 8;
  const int la = lane & 15;

  // ---- scores = (Q @ K^T) * scale ; K rows already serve as N-x-K "B" ----
  for (int t = wave; t < nt * nt; t += 8) {
    int rt = t / nt, ct = t % nt;
    v8f acc = {};
#pragma unroll
    for (int kk = 0; kk < DK; kk += 32) {
      Frag a, bf;
      a.q[0]  = *(const uint4*)&qs[(rt * 16 + la) * 72 + kk + koffL];
      a.q[1]  = *(const uint4*)&qs[(rt * 16 + la) * 72 + kk + koffL + 16];
      bf.q[0] = *(const uint4*)&ks[(ct * 16 + la) * 72 + kk + koffL];
      bf.q[1] = *(const uint4*)&ks[(ct * 16 + la) * 72 + kk + koffL + 16];
      acc = __builtin_amdgcn_wmma_f32_16x16x32_f16(false, a.v, false, bf.v,
                                                   (short)0, acc, false, false);
    }
    int mb = rt * 16 + ((lane >> 4) << 3);
    int n  = ct * 16 + la;
#pragma unroll
    for (int j = 0; j < 8; ++j)
      sc[(mb + j) * scs + n] = acc[j] * scale;
  }
  __syncthreads();

  // ---- masked softmax; probabilities written once as f16 into ph ----
  for (int i = wave; i < S; i += 8) {
    float mx = -3.0e38f;
    for (int j = lane; j < Spad; j += 32) {
      bool ok = (j < S) && ((j < maskIT) || (j <= i));
      float v = ok ? sc[i * scs + j] : -1.0e9f;
      sc[i * scs + j] = v;
      mx = fmaxf(mx, v);
    }
    for (int o = 16; o; o >>= 1) mx = fmaxf(mx, __shfl_xor(mx, o, 32));
    float sum = 0.f;
    for (int j = lane; j < Spad; j += 32) {
      float e = expf(sc[i * scs + j] - mx);
      sc[i * scs + j] = e;
      sum += e;
    }
    for (int o = 16; o; o >>= 1) sum += __shfl_xor(sum, o, 32);
    float inv = 1.0f / sum;
    for (int j = lane; j < Spad; j += 32)
      ph[i * phs + j] = (_Float16)(sc[i * scs + j] * inv);
  }
  __syncthreads();

  // ---- O = P @ V : b128 fragment loads from ph, Vt is the N-x-K "B" ----
  for (int t = wave; t < nt * 4; t += 8) {
    int rt = t >> 2, ct = t & 3;
    v8f acc = {};
    for (int k0 = 0; k0 < Spad; k0 += 32) {
      Frag a, bf;
      a.q[0]  = *(const uint4*)&ph[(rt * 16 + la) * phs + k0 + koffL];
      a.q[1]  = *(const uint4*)&ph[(rt * 16 + la) * phs + k0 + koffL + 16];
      bf.q[0] = *(const uint4*)&vt[(ct * 16 + la) * vts + k0 + koffL];
      bf.q[1] = *(const uint4*)&vt[(ct * 16 + la) * vts + k0 + koffL + 16];
      acc = __builtin_amdgcn_wmma_f32_16x16x32_f16(false, a.v, false, bf.v,
                                                   (short)0, acc, false, false);
    }
    int mb = rt * 16 + ((lane >> 4) << 3);
    int n  = ct * 16 + la;
#pragma unroll
    for (int j = 0; j < 8; ++j) {
      int m = mb + j;
      if (m < S) Om[base + (size_t)m * D_MODEL + n] = acc[j];
    }
  }
}

// =================================================================
// LayerNorm (ddof=1, matches reference): y = a*(x-mu)/(sd+eps)+b
// =================================================================
__global__ __launch_bounds__(256) void ln_kernel(
    const float* __restrict__ X, float* __restrict__ Y,
    const float* __restrict__ ga, const float* __restrict__ gb, float eps)
{
  const int row = blockIdx.x;
  const float* x = X + (size_t)row * D_MODEL;
  float* y = Y + (size_t)row * D_MODEL;
  __shared__ float red0[8], red1[8];
  float s = 0.f, ss = 0.f;
  for (int d = threadIdx.x; d < D_MODEL; d += 256) { float v = x[d]; s += v; ss += v * v; }
  for (int o = 16; o; o >>= 1) { s += __shfl_xor(s, o, 32); ss += __shfl_xor(ss, o, 32); }
  int wv = threadIdx.x >> 5, ln = threadIdx.x & 31;
  if (ln == 0) { red0[wv] = s; red1[wv] = ss; }
  __syncthreads();
  if (threadIdx.x == 0) {
    float ts = 0.f, tss = 0.f;
    for (int w = 0; w < 8; ++w) { ts += red0[w]; tss += red1[w]; }
    red0[0] = ts; red1[0] = tss;
  }
  __syncthreads();
  float mu  = red0[0] / (float)D_MODEL;
  float var = (red1[0] - (float)D_MODEL * mu * mu) / (float)(D_MODEL - 1);
  float inv = 1.0f / (sqrtf(fmaxf(var, 0.0f)) + eps);
  for (int d = threadIdx.x; d < D_MODEL; d += 256)
    y[d] = ga[d] * (x[d] - mu) * inv + gb[d];
}

// =================================================================
// Positional encoding + LayerNorm(pos_ln, eps=1e-12).
// mode 0: word_emb[cap]*sqrt(D) + tpe + PE(r)   (text)
// mode 1: src + tpe + PE(r)                     (img)
// mode 2: word_emb[CLS=1] + tpe                 (cls)
// =================================================================
__global__ __launch_bounds__(256) void posenc_kernel(
    const float* __restrict__ src, const int* __restrict__ cap,
    const float* __restrict__ word_emb, const float* __restrict__ type_pe,
    const float* __restrict__ lna, const float* __restrict__ lnb,
    float* __restrict__ dst, int type_idx, int dst_stride, int dst_off,
    int nrows, int mode)
{
  const int r = blockIdx.x, b = blockIdx.y;
  __shared__ float rowbuf[D_MODEL];
  __shared__ float red0[8], red1[8];
  for (int d = threadIdx.x; d < D_MODEL; d += 256) {
    float v;
    if (mode == 0) {
      int tok = cap[b * nrows + r];
      v = word_emb[(size_t)tok * D_MODEL + d] * SQRT_D;
    } else if (mode == 1) {
      v = src[((size_t)b * nrows + r) * D_MODEL + d];
    } else {
      v = word_emb[(size_t)1 * D_MODEL + d];
    }
    v += type_pe[type_idx * D_MODEL + d];
    if (mode != 2) {
      float ang = (float)r * expf(-(2.0f * (float)d / (float)D_MODEL) * LN10K);
      v += (d & 1) ? cosf(ang) : sinf(ang);
    }
    rowbuf[d] = v;
  }
  __syncthreads();
  float s = 0.f, ss = 0.f;
  for (int d = threadIdx.x; d < D_MODEL; d += 256) { float v = rowbuf[d]; s += v; ss += v * v; }
  for (int o = 16; o; o >>= 1) { s += __shfl_xor(s, o, 32); ss += __shfl_xor(ss, o, 32); }
  int wv = threadIdx.x >> 5, ln = threadIdx.x & 31;
  if (ln == 0) { red0[wv] = s; red1[wv] = ss; }
  __syncthreads();
  if (threadIdx.x == 0) {
    float ts = 0.f, tss = 0.f;
    for (int w = 0; w < 8; ++w) { ts += red0[w]; tss += red1[w]; }
    red0[0] = ts; red1[0] = tss;
  }
  __syncthreads();
  float mu  = red0[0] / (float)D_MODEL;
  float var = (red1[0] - (float)D_MODEL * mu * mu) / (float)(D_MODEL - 1);
  float inv = 1.0f / (sqrtf(fmaxf(var, 0.0f)) + EPS_POS);
  float* out = dst + ((size_t)b * dst_stride + dst_off + r) * D_MODEL;
  for (int d = threadIdx.x; d < D_MODEL; d += 256)
    out[d] = lna[d] * (rowbuf[d] - mu) * inv + lnb[d];
}

// row copy for concat / slice (vectorized)
__global__ __launch_bounds__(256) void copy_rows_kernel(
    const float* __restrict__ src, int src_stride, int src_off,
    float* __restrict__ dst, int dst_stride, int dst_off)
{
  const int r = blockIdx.x, b = blockIdx.y;
  const float* s = src + ((size_t)b * src_stride + src_off + r) * D_MODEL;
  float*       d = dst + ((size_t)b * dst_stride + dst_off + r) * D_MODEL;
  for (int i = threadIdx.x * 4; i < D_MODEL; i += 1024)
    *(float4*)&d[i] = *(const float4*)&s[i];
}

// =================================================================
// Host side
// =================================================================
struct LayerP {
  const float *qw,*qb,*kw,*kb,*vw,*vb,*ow,*ob;
  const float *f1w,*f1b,*f2w,*f2b,*ln1a,*ln1b,*ln2a,*ln2b;
};

extern "C" void kernel_launch(void* const* d_in, const int* in_sizes, int n_in,
                              void* d_out, int out_size, void* d_ws, size_t ws_size,
                              hipStream_t stream)
{
  (void)out_size; (void)ws_size; (void)n_in;
  auto F = [&](int i) -> const float* { return (const float*)d_in[i]; };

  const float *img1, *img2;
  const int* cap;
  const float *word_emb, *img_proj_w, *img_proj_b, *type_pe, *pos_ln_a, *pos_ln_b;
  const float *se_na, *se_nb, *de_na, *de_nb, *enc_na, *enc_nb, *out_w, *out_b;
  LayerP se[3], de[3], enc[6];

  // disambiguate flattening order by first input size:
  // insertion order  -> d_in[0] = img1 (32*49*2048)
  // jax pytree sort  -> d_in[0] = cap  (32*60)
  bool alpha = (in_sizes[0] == BATCHN * LCAP);
  int i = 0;
  if (!alpha) {
    img1 = F(i++); img2 = F(i++);
    cap = (const int*)d_in[i++]; i++;  // cap_label unused
    word_emb = F(i++); img_proj_w = F(i++); img_proj_b = F(i++); type_pe = F(i++);
    pos_ln_a = F(i++); pos_ln_b = F(i++);
    se_na = F(i++); se_nb = F(i++); de_na = F(i++); de_nb = F(i++);
    enc_na = F(i++); enc_nb = F(i++);
    auto rd = [&](LayerP& L) {
      L.qw=F(i++); L.qb=F(i++); L.kw=F(i++); L.kb=F(i++);
      L.vw=F(i++); L.vb=F(i++); L.ow=F(i++); L.ob=F(i++);
      L.f1w=F(i++); L.f1b=F(i++); L.f2w=F(i++); L.f2b=F(i++);
      L.ln1a=F(i++); L.ln1b=F(i++); L.ln2a=F(i++); L.ln2b=F(i++);
    };
    for (int l = 0; l < 3; ++l) rd(se[l]);
    for (int l = 0; l < 3; ++l) rd(de[l]);
    for (int l = 0; l < 6; ++l) rd(enc[l]);
    out_w = F(i++); out_b = F(i++);
  } else {
    cap = (const int*)d_in[i++]; i++;  // cap, cap_label
    img1 = F(i++); img2 = F(i++);
    auto rd = [&](LayerP& L) {  // alphabetical layer keys
      L.f1b=F(i++); L.f1w=F(i++); L.f2b=F(i++); L.f2w=F(i++);
      L.kb=F(i++);  L.kw=F(i++);
      L.ln1a=F(i++); L.ln1b=F(i++); L.ln2a=F(i++); L.ln2b=F(i++);
      L.ob=F(i++);  L.ow=F(i++);  L.qb=F(i++);  L.qw=F(i++);
      L.vb=F(i++);  L.vw=F(i++);
    };
    for (int l = 0; l < 3; ++l) rd(de[l]);
    de_na = F(i++); de_nb = F(i++);
    for (int l = 0; l < 6; ++l) rd(enc[l]);
    enc_na = F(i++); enc_nb = F(i++);
    img_proj_b = F(i++); img_proj_w = F(i++);
    out_b = F(i++); out_w = F(i++);
    pos_ln_a = F(i++); pos_ln_b = F(i++);
    for (int l = 0; l < 3; ++l) rd(se[l]);
    se_na = F(i++); se_nb = F(i++);
    type_pe = F(i++); word_emb = F(i++);
  }

  // ---- workspace carve (f32) ----
  float* wp = (float*)d_ws;
  auto alloc = [&](size_t n) { float* r = wp; wp += n; return r; };
  const size_t RMAX = (size_t)BATCHN * 160;
  float* LN1 = alloc(RMAX * D_MODEL);
  float* QB  = alloc(RMAX * D_MODEL);
  float* KB  = alloc(RMAX * D_MODEL);
  float* VB  = alloc(RMAX * D_MODEL);
  float* AO  = alloc(RMAX * D_MODEL);
  float* FFB = alloc(RMAX * FF_DIM);
  float* X1  = alloc((size_t)BATCHN * 50  * D_MODEL);
  float* X2  = alloc((size_t)BATCHN * 50  * D_MODEL);
  float* XI  = alloc((size_t)BATCHN * 100 * D_MODEL);
  float* XT  = alloc((size_t)BATCHN * LCAP * D_MODEL);
  float* XE  = alloc((size_t)BATCHN * 160 * D_MODEL);
  float* PRJ = alloc((size_t)BATCHN * NPATCH * D_MODEL);
  float* TXT = alloc((size_t)BATCHN * LCAP * D_MODEL);

  auto run_gemm = [&](const float* A, const float* W, const float* bias,
                      const float* res, float* C, int M, int N, int K, int act) {
    dim3 g((N + BN - 1) / BN, (M + BM - 1) / BM);
    gemm_kernel<<<g, 256, 0, stream>>>(A, W, bias, res, C, M, N, K, act);
  };
  auto run_ln = [&](const float* X, float* Y, const float* a, const float* b, int rows) {
    ln_kernel<<<rows, 256, 0, stream>>>(X, Y, a, b, EPS_LN);
  };
  auto run_attn = [&](int S, int maskIT) {
    int Spad = (S + 15) & ~15;
    size_t sm = (size_t)Spad * 72 * sizeof(_Float16) * 2            // qs, ks
              + (size_t)64 * (Spad + 8) * sizeof(_Float16)          // vt
              + (size_t)Spad * (Spad + 8) * sizeof(_Float16)        // ph
              + (size_t)Spad * (Spad + 4) * sizeof(float);          // sc
    attn_kernel<<<dim3(NHEAD, BATCHN), 256, sm, stream>>>(QB, KB, VB, AO,
                                                          S, Spad, maskIT, 0.125f);
  };
  auto run_stage = [&](float* x, int S, const float* na, const float* nb,
                       const LayerP* Ls, int nl, int maskIT) {
    int rows = BATCHN * S;
    run_ln(x, x, na, nb, rows);
    for (int li = 0; li < nl; ++li) {
      const LayerP& p = Ls[li];
      run_ln(x, LN1, p.ln1a, p.ln1b, rows);
      run_gemm(LN1, p.qw, p.qb, nullptr, QB, rows, D_MODEL, D_MODEL, 0);
      run_gemm(LN1, p.kw, p.kb, nullptr, KB, rows, D_MODEL, D_MODEL, 0);
      run_gemm(LN1, p.vw, p.vb, nullptr, VB, rows, D_MODEL, D_MODEL, 0);
      run_attn(S, maskIT);
      run_gemm(AO, p.ow, p.ob, x, x, rows, D_MODEL, D_MODEL, 0);
      run_ln(x, LN1, p.ln2a, p.ln2b, rows);
      run_gemm(LN1, p.f1w, p.f1b, nullptr, FFB, rows, FF_DIM, D_MODEL, 1);
      run_gemm(FFB, p.f2w, p.f2b, x, x, rows, D_MODEL, FF_DIM, 0);
    }
  };

  // raise dynamic-LDS cap for the attention kernel (harmless if no-op)
  (void)hipFuncSetAttribute((const void*)attn_kernel,
                            hipFuncAttributeMaxDynamicSharedMemorySize, 229376);

  // ---------------- pipeline ----------------
  // text embeddings -> XT
  posenc_kernel<<<dim3(LCAP, BATCHN), 256, 0, stream>>>(
      nullptr, cap, word_emb, type_pe, pos_ln_a, pos_ln_b, XT, 6, LCAP, 0, LCAP, 0);

  // image stream 1: sigmoid(img1 @ Wp + bp) -> PRJ; posenc -> X1[1..49]; cls -> X1[0]
  run_gemm(img1, img_proj_w, img_proj_b, nullptr, PRJ,
           BATCHN * NPATCH, D_MODEL, 2048, 2);
  posenc_kernel<<<dim3(NPATCH, BATCHN), 256, 0, stream>>>(
      PRJ, nullptr, word_emb, type_pe, pos_ln_a, pos_ln_b, X1, 2, 50, 1, NPATCH, 1);
  posenc_kernel<<<dim3(1, BATCHN), 256, 0, stream>>>(
      nullptr, nullptr, word_emb, type_pe, pos_ln_a, pos_ln_b, X1, 1, 50, 0, 1, 2);

  // image stream 2
  run_gemm(img2, img_proj_w, img_proj_b, nullptr, PRJ,
           BATCHN * NPATCH, D_MODEL, 2048, 2);
  posenc_kernel<<<dim3(NPATCH, BATCHN), 256, 0, stream>>>(
      PRJ, nullptr, word_emb, type_pe, pos_ln_a, pos_ln_b, X2, 4, 50, 1, NPATCH, 1);
  posenc_kernel<<<dim3(1, BATCHN), 256, 0, stream>>>(
      nullptr, nullptr, word_emb, type_pe, pos_ln_a, pos_ln_b, X2, 3, 50, 0, 1, 2);

  // single-stream encoders (shared weights) on each image stream
  run_stage(X1, 50, se_na, se_nb, se, 3, NOMASK);
  run_stage(X2, 50, se_na, se_nb, se, 3, NOMASK);

  // concat -> XI, dual encoder
  copy_rows_kernel<<<dim3(50, BATCHN), 256, 0, stream>>>(X1, 50, 0, XI, 100, 0);
  copy_rows_kernel<<<dim3(50, BATCHN), 256, 0, stream>>>(X2, 50, 0, XI, 100, 50);
  run_stage(XI, 100, de_na, de_nb, de, 3, NOMASK);

  // concat img+text -> XE, masked encoder (img tokens fully visible, text causal)
  copy_rows_kernel<<<dim3(100, BATCHN), 256, 0, stream>>>(XI, 100, 0, XE, 160, 0);
  copy_rows_kernel<<<dim3(LCAP, BATCHN), 256, 0, stream>>>(XT, LCAP, 0, XE, 160, 100);
  run_stage(XE, 160, enc_na, enc_nb, enc, 6, 100);

  // slice text tokens and project to vocab
  copy_rows_kernel<<<dim3(LCAP, BATCHN), 256, 0, stream>>>(XE, 160, 100, TXT, LCAP, 0);
  run_gemm(TXT, out_w, out_b, nullptr, (float*)d_out,
           BATCHN * LCAP, VOCAB, D_MODEL, 0);
}
